// GAT_47957604827269
// MI455X (gfx1250) — matrix operand order repaired and checked by reading, not compiled, not run on hardware
//
#include <hip/hip_runtime.h>
#include <stdint.h>

// ---------------- CDNA5 WMMA types ----------------
typedef __attribute__((ext_vector_type(16))) __bf16 v16bf;
typedef __attribute__((ext_vector_type(8)))  float  v8f;

// ---------------- helpers ----------------
__device__ __forceinline__ unsigned short f2bf(float f) {
    unsigned u = __builtin_bit_cast(unsigned, f);
    u += 0x7FFFu + ((u >> 16) & 1u);          // round-to-nearest-even
    return (unsigned short)(u >> 16);
}
// order-preserving float <-> uint encoding for atomicMax-based segment max
__device__ __forceinline__ unsigned encf(float f) {
    unsigned u = __builtin_bit_cast(unsigned, f);
    return (u & 0x80000000u) ? ~u : (u | 0x80000000u);
}
__device__ __forceinline__ float decf(unsigned e) {
    unsigned u = (e & 0x80000000u) ? (e & 0x7FFFFFFFu) : ~e;
    return __builtin_bit_cast(float, u);
}
#define ENC_NEG_INF 0x007FFFFFu   // encf(-inf)

// ---------------- utility kernels ----------------
__global__ void k_fillf(float* p, float v, long long n) {
    long long i = (long long)blockIdx.x * blockDim.x + threadIdx.x;
    if (i < n) p[i] = v;
}
__global__ void k_fillu(unsigned* p, unsigned v, long long n) {
    long long i = (long long)blockIdx.x * blockDim.x + threadIdx.x;
    if (i < n) p[i] = v;
}
__global__ void k_f2bf(const float* __restrict__ s, unsigned short* __restrict__ d, long long n) {
    long long i = (long long)blockIdx.x * blockDim.x + threadIdx.x;
    if (i < n) d[i] = f2bf(s[i]);
}

// Pack W [K x NCOLS] fp32 row-major into per-lane WMMA B fragments (bf16).
// Layout consumed by GEMM: Bp[(((kc*NCT + ctg)*32 + lane)*16 + e]
//   lane 0-15  : col = ctg*16 + lane, K = kc*32 + (e<8 ? e : e+8)        (e = 0..15)
//   lane 16-31 : col = ctg*16 + lane-16, K = kc*32 + 8 + (e<8 ? e : e+8)
__global__ void k_packB(const float* __restrict__ W, unsigned short* __restrict__ Wp,
                        int K, int NCOLS) {
    int NCT = NCOLS >> 4;
    int KC  = K >> 5;
    int t = blockIdx.x * blockDim.x + threadIdx.x;
    if (t >= KC * NCT * 32) return;
    int lane = t & 31;
    int idx  = t >> 5;
    int ctg  = idx % NCT;
    int kc   = idx / NCT;
    int col  = ctg * 16 + (lane & 15);
    int kb   = kc * 32 + ((lane >> 4) * 8);
    unsigned short* out = Wp + (size_t)t * 16;
#pragma unroll
    for (int e = 0; e < 16; ++e)
        out[e] = f2bf(W[(size_t)(kb + (e < 8 ? e : e + 8)) * NCOLS + col]);
}

// ---------------- WMMA bf16 GEMM: C[nrows x NCOLS] = A[nrows x K] * B ----------------
// One wave -> 16 rows x 64 cols (4 WMMA accumulators). blockDim = 256 (8 waves).
// grid.x covers row strips, grid.y covers 64-column blocks.
__global__ void __launch_bounds__(256) k_gemm_bf16(
    const unsigned short* __restrict__ A,   // bf16 row-major [nrows x K]
    const unsigned short* __restrict__ Bp,  // packed fragments
    float* __restrict__ C,                  // fp32 row-major [nrows x NCOLS]
    int nrows, int K, int NCOLS) {

    int lane = threadIdx.x & 31;
    int wave = threadIdx.x >> 5;
    int gw   = blockIdx.x * 8 + wave;
    int row0 = gw * 16;
    if (row0 >= nrows) return;

    int NCT = NCOLS >> 4;
    int nkc = K >> 5;
    int ct0 = blockIdx.y * 4;

    int arow = row0 + (lane & 15);
    if (arow >= nrows) arow = nrows - 1;     // tail guard (reads only)
    int asub = (lane >> 4) * 8;              // A fragment: K-subgroup per half-wave

    union { uint4 q[2]; v16bf v; } af, bfr;
    v8f acc[4] = {};

    for (int kc = 0; kc < nkc; ++kc) {
        const unsigned short* ap = A + (size_t)arow * K + kc * 32 + asub;
        af.q[0] = *(const uint4*)(const void*)(ap);        // K = kc*32+asub   .. +7
        af.q[1] = *(const uint4*)(const void*)(ap + 16);   // K = kc*32+asub+16.. +23
#pragma unroll
        for (int t = 0; t < 4; ++t) {
            const unsigned short* bp =
                Bp + ((size_t)(kc * NCT + ct0 + t) * 32 + lane) * 16;
            bfr.q[0] = *(const uint4*)(const void*)(bp);
            bfr.q[1] = *(const uint4*)(const void*)(bp + 8);
            acc[t] = __builtin_amdgcn_wmma_f32_16x16x32_bf16(
                false, af.v, false, bfr.v, (short)0, acc[t], false, false);
        }
    }

    // C/D layout: VGPR r -> (M = (lane>=16 ? 8 : 0)+r, N = lane&15)
    int mbase = row0 + asub;
    int cb    = blockIdx.y * 64 + (lane & 15);
#pragma unroll
    for (int t = 0; t < 4; ++t)
#pragma unroll
        for (int r = 0; r < 8; ++r) {
            int m = mbase + r;
            if (m < nrows) C[(size_t)m * NCOLS + cb + t * 16] = acc[t][r];
        }
}

// ---------------- attention coefficients ----------------
__global__ void k_alpha(const float* __restrict__ Hm, const float* __restrict__ av,
                        const float* __restrict__ dv, float* __restrict__ outs,
                        float* __restrict__ outd, int N, int H, int Cc) {
    int t = blockIdx.x * blockDim.x + threadIdx.x;
    if (t >= N * H) return;
    int n = t / H, h = t % H;
    const float* hp = Hm + (size_t)n * H * Cc + (size_t)h * Cc;
    const float* a1 = av + h * Cc;
    const float* a2 = dv + h * Cc;
    float s = 0.f, d = 0.f;
    for (int c = 0; c < Cc; ++c) { float v = hp[c]; s += v * a1[c]; d += v * a2[c]; }
    outs[t] = s; outd[t] = d;
}

// ---------------- edge pass 1: leaky-relu(e) and scatter-max into dst ----------------
__global__ void k_edge_max(const int* __restrict__ esrc, const int* __restrict__ edst,
                           const float* __restrict__ as, const float* __restrict__ ad,
                           unsigned* __restrict__ menc, float* __restrict__ ebuf,
                           int E, int N, int H) {
    int t = blockIdx.x * blockDim.x + threadIdx.x;
    if (t >= E + N) return;
    int s = (t < E) ? esrc[t] : (t - E);
    int d = (t < E) ? edst[t] : (t - E);
    for (int h = 0; h < H; ++h) {
        float e = as[(size_t)s * H + h] + ad[(size_t)d * H + h];
        e = (e > 0.f) ? e : 0.2f * e;                  // leaky_relu, slope 0.2
        ebuf[(size_t)t * H + h] = e;
        atomicMax(&menc[(size_t)d * H + h], encf(e));
    }
}

// ---------------- edge pass 2: exp(e - m[dst]) and scatter-sum ----------------
__global__ void k_edge_exp(const int* __restrict__ esrc, const int* __restrict__ edst,
                           const unsigned* __restrict__ menc, float* __restrict__ ebuf,
                           float* __restrict__ ssum, int E, int N, int H) {
    int t = blockIdx.x * blockDim.x + threadIdx.x;
    if (t >= E + N) return;
    int d = (t < E) ? edst[t] : (t - E);
    for (int h = 0; h < H; ++h) {
        float m  = decf(menc[(size_t)d * H + h]);
        float ex = expf(ebuf[(size_t)t * H + h] - m);
        ebuf[(size_t)t * H + h] = ex;
        atomicAdd(&ssum[(size_t)d * H + h], ex);
    }
}

// ---------------- edge pass 3: weighted scatter aggregation ----------------
// one thread per (edge, head, 4 channels)
__global__ void k_edge_agg(const int* __restrict__ esrc, const int* __restrict__ edst,
                           const float* __restrict__ ebuf, const float* __restrict__ ssum,
                           const float* __restrict__ Hm, float* __restrict__ acc,
                           int E, int N, int H, int Cc) {
    long long t = (long long)blockIdx.x * blockDim.x + threadIdx.x;
    int per = H * (Cc >> 2);
    long long total = (long long)(E + N) * per;
    if (t >= total) return;
    int eix = (int)(t / per);
    int rem = (int)(t % per);
    int h   = rem / (Cc >> 2);
    int c4  = (rem % (Cc >> 2)) * 4;
    int s = (eix < E) ? esrc[eix] : (eix - E);
    int d = (eix < E) ? edst[eix] : (eix - E);
    float alpha = ebuf[(size_t)eix * H + h] / (ssum[(size_t)d * H + h] + 1e-16f);
    const float* hp = Hm  + (size_t)s * H * Cc + (size_t)h * Cc + c4;
    float*       ap = acc + (size_t)d * H * Cc + (size_t)h * Cc + c4;
    atomicAdd(ap + 0, alpha * hp[0]);
    atomicAdd(ap + 1, alpha * hp[1]);
    atomicAdd(ap + 2, alpha * hp[2]);
    atomicAdd(ap + 3, alpha * hp[3]);
}

// ---------------- layer-0 epilogue: +bias, ELU, convert to bf16 for GEMM1 ----------------
__global__ void k_final0(const float* __restrict__ acc, const float* __restrict__ bias,
                         unsigned short* __restrict__ a1b, long long total) {
    long long t = (long long)blockIdx.x * blockDim.x + threadIdx.x;
    if (t >= total) return;
    int j = (int)(t & 255);
    float v = acc[t] + bias[j];
    v = (v > 0.f) ? v : expm1f(v);                 // ELU(alpha=1)
    a1b[t] = f2bf(v);
}

// ---------------- layer-1 epilogue: +bias -> d_out ----------------
__global__ void k_final1(const float* __restrict__ acc, const float* __restrict__ bias,
                         float* __restrict__ out, long long total) {
    long long t = (long long)blockIdx.x * blockDim.x + threadIdx.x;
    if (t >= total) return;
    out[t] = acc[t] + bias[(int)(t & 63)];
}

// =====================================================================
extern "C" void kernel_launch(void* const* d_in, const int* in_sizes, int n_in,
                              void* d_out, int out_size, void* d_ws, size_t ws_size,
                              hipStream_t stream) {
    const float* x    = (const float*)d_in[0];
    const int*   ei   = (const int*)  d_in[1];
    const float* W0   = (const float*)d_in[2];
    const float* as0w = (const float*)d_in[3];
    const float* ad0w = (const float*)d_in[4];
    const float* b0   = (const float*)d_in[5];
    const float* W1   = (const float*)d_in[6];
    const float* as1w = (const float*)d_in[7];
    const float* ad1w = (const float*)d_in[8];
    const float* b1   = (const float*)d_in[9];
    float* out = (float*)d_out;

    const int FIN = 128, H0 = 4, HID = 64, HC0 = 256;
    const int N = in_sizes[0] / FIN;
    const int E = in_sizes[1] / 2;
    const int Etot = E + N;
    const int* esrc = ei;
    const int* edst = ei + E;

    // ---- workspace sub-allocation (with aliasing for layer 1) ----
    char* w = (char*)d_ws;
    size_t off = 0;
    auto take = [&](size_t bytes) -> char* {
        char* p = w + off;
        off += (bytes + 255) & ~(size_t)255;
        return p;
    };
    unsigned short* xb  = (unsigned short*)take((size_t)N * FIN * 2);
    unsigned short* w0p = (unsigned short*)take((size_t)FIN * HC0 * 2);
    unsigned short* w1p = (unsigned short*)take((size_t)HC0 * HID * 2);
    char*  h0reg = take((size_t)N * HC0 * 4);   // h0; later a1b + h1
    float* h0    = (float*)h0reg;
    float*    as0 = (float*)   take((size_t)N * H0 * 4);
    float*    ad0 = (float*)   take((size_t)N * H0 * 4);
    unsigned* m0  = (unsigned*)take((size_t)N * H0 * 4);
    float*    s0  = (float*)   take((size_t)N * H0 * 4);
    float*    ex0 = (float*)   take((size_t)Etot * H0 * 4);
    char*  accreg = take((size_t)N * HC0 * 4);  // acc0; later ex1/acc1/etc
    float* acc0   = (float*)accreg;
    // phase-2 aliases (safe: old contents dead by the time these are written)
    unsigned short* a1b = (unsigned short*)h0reg;               // N*256 bf16
    float* h1 = (float*)(h0reg + (size_t)N * HC0 * 2);          // N*64 f32
    size_t ex1b  = ((size_t)Etot * 4 + 255) & ~(size_t)255;
    size_t acc1b = ((size_t)N * HID * 4 + 255) & ~(size_t)255;
    float*    ex1  = (float*)accreg;
    float*    acc1 = (float*)(accreg + ex1b);
    float*    as1  = (float*)(accreg + ex1b + acc1b);
    float*    ad1  = as1 + N;
    unsigned* m1   = (unsigned*)(ad1 + N);
    float*    s1   = (float*)(m1 + N);
    (void)ws_size; (void)n_in; (void)out_size;

    const int TB = 256;
    auto gr = [&](long long n) { return dim3((unsigned)((n + TB - 1) / TB)); };

    // ---- prep: conversions + weight packing ----
    k_f2bf <<<gr((long long)N * FIN), TB, 0, stream>>>(x, xb, (long long)N * FIN);
    k_packB<<<gr((FIN / 32) * (HC0 / 16) * 32), TB, 0, stream>>>(W0, w0p, FIN, HC0);
    k_packB<<<gr((HC0 / 32) * (HID / 16) * 32), TB, 0, stream>>>(W1, w1p, HC0, HID);

    // ---- layer 0 ----
    k_fillf<<<gr((long long)N * HC0), TB, 0, stream>>>(acc0, 0.f, (long long)N * HC0);
    k_fillf<<<gr((long long)N * H0),  TB, 0, stream>>>(s0, 0.f, (long long)N * H0);
    k_fillu<<<gr((long long)N * H0),  TB, 0, stream>>>(m0, ENC_NEG_INF, (long long)N * H0);

    int nwaves = (N + 15) / 16;
    dim3 g0((nwaves + 7) / 8, HC0 / 64);
    k_gemm_bf16<<<g0, 256, 0, stream>>>(xb, w0p, h0, N, FIN, HC0);

    k_alpha   <<<gr((long long)N * H0), TB, 0, stream>>>(h0, as0w, ad0w, as0, ad0, N, H0, HID);
    k_edge_max<<<gr(Etot), TB, 0, stream>>>(esrc, edst, as0, ad0, m0, ex0, E, N, H0);
    k_edge_exp<<<gr(Etot), TB, 0, stream>>>(esrc, edst, m0, ex0, s0, E, N, H0);
    k_edge_agg<<<gr((long long)Etot * H0 * (HID / 4)), TB, 0, stream>>>(
        esrc, edst, ex0, s0, h0, acc0, E, N, H0, HID);
    k_final0<<<gr((long long)N * HC0), TB, 0, stream>>>(acc0, b0, a1b, (long long)N * HC0);

    // ---- layer 1 ----
    k_fillf<<<gr((long long)N * HID), TB, 0, stream>>>(acc1, 0.f, (long long)N * HID);
    k_fillf<<<gr(N), TB, 0, stream>>>(s1, 0.f, N);
    k_fillu<<<gr(N), TB, 0, stream>>>(m1, ENC_NEG_INF, N);

    dim3 g1g((nwaves + 7) / 8, HID / 64);
    k_gemm_bf16<<<g1g, 256, 0, stream>>>(a1b, w1p, h1, N, HC0, HID);

    k_alpha   <<<gr(N), TB, 0, stream>>>(h1, as1w, ad1w, as1, ad1, N, 1, HID);
    k_edge_max<<<gr(Etot), TB, 0, stream>>>(esrc, edst, as1, ad1, m1, ex1, E, N, 1);
    k_edge_exp<<<gr(Etot), TB, 0, stream>>>(esrc, edst, m1, ex1, s1, E, N, 1);
    k_edge_agg<<<gr((long long)Etot * (HID / 4)), TB, 0, stream>>>(
        esrc, edst, ex1, s1, h1, acc1, E, N, 1, HID);
    k_final1<<<gr((long long)N * HID), TB, 0, stream>>>(acc1, b1, out, (long long)N * HID);
}
